// PoincareLogExpDistanceHead_55009941127534
// MI455X (gfx1250) — matrix operand order repaired and checked by reading, not compiled, not run on hardware
//
#include <hip/hip_runtime.h>
#include <math.h>

typedef __attribute__((ext_vector_type(16))) _Float16 v16h;
typedef __attribute__((ext_vector_type(8)))  float    v8f;
typedef __attribute__((ext_vector_type(4)))  unsigned int u32x4;
typedef __attribute__((ext_vector_type(4)))  float    f32x4;
typedef __attribute__((ext_vector_type(4)))  _Float16 f16x4;

#define POIN_EPS   0.01f
#define NORM_EPS_F 1e-5f
#define LDSTRIDE   40        // 32 K-halves + 8 pad halves (80 B rows)
#define BUFHALVES  (128 * LDSTRIDE)
#define NKSTEPS    32        // K = 1024, BK = 32

// ---------------------------------------------------------------- converts
__global__ void cvt_f32_to_f16(const float* __restrict__ in,
                               _Float16* __restrict__ out, int n4) {
    int i = blockIdx.x * blockDim.x + threadIdx.x;
    if (i < n4) {
        f32x4 v = ((const f32x4*)in)[i];
        f16x4 h;
        h.x = (_Float16)v.x; h.y = (_Float16)v.y;
        h.z = (_Float16)v.z; h.w = (_Float16)v.w;
        ((f16x4*)out)[i] = h;
    }
}

// ------------------------------------------------- exp-map rows + sq-norms
// out_h[row] = tanh(n)/n * in[row]   (n = max(||in[row]||, 1e-5))
// out_s2[row] = || exp-mapped row ||^2  (f32)
// H == 1024, 256 threads -> 4 elements/thread cached in registers.
__global__ __launch_bounds__(256) void expmap_rows(
        const float* __restrict__ in, _Float16* __restrict__ out_h,
        float* __restrict__ out_s2) {
    __shared__ float red[8];
    const int row = blockIdx.x, tid = threadIdx.x;
    const float* rp = in + (size_t)row * 1024;
    float v[4];
    float sq = 0.f;
#pragma unroll
    for (int k = 0; k < 4; ++k) { v[k] = rp[tid + k * 256]; sq += v[k] * v[k]; }
#pragma unroll
    for (int o = 16; o > 0; o >>= 1) sq += __shfl_xor(sq, o, 32);
    int lane = tid & 31, wid = tid >> 5;
    if (lane == 0) red[wid] = sq;
    __syncthreads();
    if (tid < 32) {
        float s = (lane < 8) ? red[lane] : 0.f;
#pragma unroll
        for (int o = 4; o > 0; o >>= 1) s += __shfl_xor(s, o, 32);
        if (lane == 0) red[0] = s;
    }
    __syncthreads();
    float nsq = red[0];
    float nc = fmaxf(sqrtf(nsq), NORM_EPS_F);
    float scale = tanhf(nc) / nc;
    if (tid == 0) out_s2[row] = scale * scale * nsq;
    _Float16* op = out_h + (size_t)row * 1024;
#pragma unroll
    for (int k = 0; k < 4; ++k) op[tid + k * 256] = (_Float16)(v[k] * scale);
}

// ------------------------------------------------------------ WMMA GEMM core
// 128x128 block tile, 256 threads = 8 waves (4M x 2N), BK=32, K=1024.
// Double-buffered LDS: one barrier per K-step; global loads for chunk k+1
// overlap the WMMAs on chunk k.
__device__ __forceinline__ void frag_mma(const _Float16* bufA,
                                         const _Float16* bufB,
                                         v8f (&acc)[2][4], int wm, int wn,
                                         int laneHalf, int laneGrp) {
    v16h af[2], bf[4];
#pragma unroll
    for (int i = 0; i < 2; ++i) {
        // A 16x32 f16: lanes 0-15 -> K0-7 / K16-23, lanes 16-31 -> K8-15 / K24-31
        const _Float16* base = bufA + (wm * 32 + i * 16 + laneHalf) * LDSTRIDE;
        _Float16* ap = (_Float16*)&af[i];
        *(u32x4*)(ap)     = *(const u32x4*)(base + laneGrp * 8);
        *(u32x4*)(ap + 8) = *(const u32x4*)(base + 16 + laneGrp * 8);
    }
#pragma unroll
    for (int j = 0; j < 4; ++j) {
        // B 32x16 f16: lane = column, lanes 0-15 -> K0-7 / K16-23, lanes 16-31 -> K8-15 / K24-31
        const _Float16* base =
            bufB + (wn * 64 + j * 16 + laneHalf) * LDSTRIDE + laneGrp * 8;
        _Float16* bp = (_Float16*)&bf[j];
        *(u32x4*)(bp)     = *(const u32x4*)(base);
        *(u32x4*)(bp + 8) = *(const u32x4*)(base + 16);
    }
#pragma unroll
    for (int i = 0; i < 2; ++i)
#pragma unroll
        for (int j = 0; j < 4; ++j)
            acc[i][j] = __builtin_amdgcn_wmma_f32_16x16x32_f16(
                false, af[i], false, bf[j], (short)0, acc[i][j], false, false);
}

__device__ __forceinline__ void gemm_core(
        const _Float16* __restrict__ A, const _Float16* __restrict__ B,
        int lda, int ldb, int row0, int col0,
        _Float16* sA, _Float16* sB, v8f (&acc)[2][4],
        int t, int wm, int wn, int laneHalf, int laneGrp) {
    const int r0 = t >> 2, c4 = t & 3;   // 2 rows/thread (r0, r0+64), 16B each
    const _Float16* gA0 = A + (size_t)(row0 + r0) * lda + c4 * 8;
    const _Float16* gA1 = gA0 + (size_t)64 * lda;
    const _Float16* gB0 = B + (size_t)(col0 + r0) * ldb + c4 * 8;
    const _Float16* gB1 = gB0 + (size_t)64 * ldb;
    _Float16* stA0 = sA + r0 * LDSTRIDE + c4 * 8;
    _Float16* stA1 = stA0 + 64 * LDSTRIDE;
    _Float16* stB0 = sB + r0 * LDSTRIDE + c4 * 8;
    _Float16* stB1 = stB0 + 64 * LDSTRIDE;

    // prologue: chunk 0 -> buffer 0
    *(u32x4*)stA0 = *(const u32x4*)gA0;  gA0 += 32;
    *(u32x4*)stA1 = *(const u32x4*)gA1;  gA1 += 32;
    *(u32x4*)stB0 = *(const u32x4*)gB0;  gB0 += 32;
    *(u32x4*)stB1 = *(const u32x4*)gB1;  gB1 += 32;
    __syncthreads();

#pragma unroll 2
    for (int i = 0; i < NKSTEPS - 1; ++i) {
        u32x4 na0 = *(const u32x4*)gA0;  gA0 += 32;
        u32x4 na1 = *(const u32x4*)gA1;  gA1 += 32;
        u32x4 nb0 = *(const u32x4*)gB0;  gB0 += 32;
        u32x4 nb1 = *(const u32x4*)gB1;  gB1 += 32;
        const int cur = (i & 1) * BUFHALVES;
        const int nxt = BUFHALVES - cur;
        frag_mma(sA + cur, sB + cur, acc, wm, wn, laneHalf, laneGrp);
        *(u32x4*)(stA0 + nxt) = na0;
        *(u32x4*)(stA1 + nxt) = na1;
        *(u32x4*)(stB0 + nxt) = nb0;
        *(u32x4*)(stB1 + nxt) = nb1;
        __syncthreads();
    }
    const int last = ((NKSTEPS - 1) & 1) * BUFHALVES;
    frag_mma(sA + last, sB + last, acc, wm, wn, laneHalf, laneGrp);
}

// ----------------------------------------------- tangent GEMM: x = hs@W^T + b
__global__ __launch_bounds__(256) void gemm_tangent(
        const _Float16* __restrict__ A, const _Float16* __restrict__ Bm,
        const float* __restrict__ bias, float* __restrict__ X,
        int lda, int ldb, int ldx) {
    __shared__ _Float16 sA[2 * BUFHALVES];
    __shared__ _Float16 sB[2 * BUFHALVES];
    const int t = threadIdx.x;
    const int row0 = blockIdx.y * 128, col0 = blockIdx.x * 128;
    const int lane = t & 31, wave = t >> 5;
    const int wm = wave >> 1, wn = wave & 1;
    const int laneHalf = lane & 15, laneGrp = lane >> 4;

    v8f acc[2][4] = {};
    gemm_core(A, Bm, lda, ldb, row0, col0, sA, sB, acc,
              t, wm, wn, laneHalf, laneGrp);

#pragma unroll
    for (int j = 0; j < 4; ++j) {
        int col = col0 + wn * 64 + j * 16 + laneHalf;
        float bv = bias[col];
#pragma unroll
        for (int i = 0; i < 2; ++i) {
            int rbase = row0 + wm * 32 + i * 16 + laneGrp * 8;
#pragma unroll
            for (int r = 0; r < 8; ++r)
                X[(size_t)(rbase + r) * ldx + col] = acc[i][j][r] + bv;
        }
    }
}

// ---------------------- distance GEMM: out = -arccosh(1 + 2 d2 / (denom+eps))
__global__ __launch_bounds__(256) void gemm_dist(
        const _Float16* __restrict__ A, const _Float16* __restrict__ Bm,
        const float* __restrict__ x2, const float* __restrict__ y2,
        float* __restrict__ out, int lda, int ldb, int V) {
    __shared__ _Float16 sA[2 * BUFHALVES];
    __shared__ _Float16 sB[2 * BUFHALVES];
    const int t = threadIdx.x;
    const int row0 = blockIdx.y * 128, col0 = blockIdx.x * 128;
    const int lane = t & 31, wave = t >> 5;
    const int wm = wave >> 1, wn = wave & 1;
    const int laneHalf = lane & 15, laneGrp = lane >> 4;

    v8f acc[2][4] = {};
    gemm_core(A, Bm, lda, ldb, row0, col0, sA, sB, acc,
              t, wm, wn, laneHalf, laneGrp);

    float x2l[2][8];
#pragma unroll
    for (int i = 0; i < 2; ++i)
#pragma unroll
        for (int r = 0; r < 8; ++r)
            x2l[i][r] = x2[row0 + wm * 32 + i * 16 + laneGrp * 8 + r];

#pragma unroll
    for (int j = 0; j < 4; ++j) {
        int col = col0 + wn * 64 + j * 16 + laneHalf;
        float y2v = y2[col];
        float omy = 1.f - y2v;
#pragma unroll
        for (int i = 0; i < 2; ++i) {
            int rbase = row0 + wm * 32 + i * 16 + laneGrp * 8;
#pragma unroll
            for (int r = 0; r < 8; ++r) {
                float x2v = x2l[i][r];
                float g = acc[i][j][r];
                float d2 = fmaxf(x2v + y2v - 2.f * g, 0.f);
                float denom = (1.f - x2v) * omy + POIN_EPS;
                float arg = fmaxf(1.f + __fdividef(2.f * d2, denom),
                                  1.f + POIN_EPS);
                float dist = __logf(arg + sqrtf(arg * arg - 1.f));
                out[(size_t)(rbase + r) * V + col] = -dist;
            }
        }
    }
}

// ----------------------------------------------------------------- launcher
extern "C" void kernel_launch(void* const* d_in, const int* in_sizes, int n_in,
                              void* d_out, int out_size, void* d_ws,
                              size_t ws_size, hipStream_t stream) {
    const float* hs = (const float*)d_in[0];   // (2,1024,1024)
    const float* W  = (const float*)d_in[1];   // (1024,1024)
    const float* b  = (const float*)d_in[2];   // (1024,)
    const float* wt = (const float*)d_in[3];   // (32000,1024)
    float* out = (float*)d_out;                // (2,1024,32000)

    const int H = 1024, V = 32000, N = 2048;   // N = B*S

    char* ws = (char*)d_ws;
    size_t off = 0;
    _Float16* hs_h = (_Float16*)(ws + off); off += (size_t)N * H * 2;
    _Float16* W_h  = (_Float16*)(ws + off); off += (size_t)H * H * 2;
    float*    x    = (float*)(ws + off);    off += (size_t)N * H * 4;
    _Float16* xe_h = (_Float16*)(ws + off); off += (size_t)N * H * 2;
    float*    x2   = (float*)(ws + off);    off += (size_t)N * 4;
    float*    y2   = (float*)(ws + off);    off += (size_t)V * 4;
    _Float16* ye_h = (_Float16*)(ws + off); off += (size_t)V * H * 2;

    // 1. f32 -> f16 converts for the tangent GEMM operands
    cvt_f32_to_f16<<<(N * H / 4 + 255) / 256, 256, 0, stream>>>(hs, hs_h, N * H / 4);
    cvt_f32_to_f16<<<(H * H / 4 + 255) / 256, 256, 0, stream>>>(W, W_h, H * H / 4);

    // 2. x = hs @ W^T + b   (2048 x 1024, K=1024)  [WMMA, double-buffered]
    gemm_tangent<<<dim3(H / 128, N / 128), 256, 0, stream>>>(
        hs_h, W_h, b, x, H, H, H);

    // 3. exp-map rows -> f16 operands + squared norms
    expmap_rows<<<N, 256, 0, stream>>>(x, xe_h, x2);
    expmap_rows<<<V, 256, 0, stream>>>(wt, ye_h, y2);

    // 4. distances: -arccosh over xe @ ye^T (2048 x 32000, K=1024)  [WMMA]
    gemm_dist<<<dim3(V / 128, N / 128), 256, 0, stream>>>(
        xe_h, ye_h, x2, y2, out, H, H, V);
}
